// SAMN_50294067036843
// MI455X (gfx1250) — matrix operand order repaired and checked by reading, not compiled, not run on hardware
//
#include <hip/hip_runtime.h>

#define Bsz   2048
#define Fdim  512
#define MFdim 4096
#define Cn    20

typedef __attribute__((ext_vector_type(16))) _Float16 v16h;
typedef __attribute__((ext_vector_type(8)))  _Float16 h8;
typedef __attribute__((ext_vector_type(4)))  _Float16 h4;
typedef __attribute__((ext_vector_type(8)))  float    v8f;
typedef __attribute__((ext_vector_type(4)))  float    f4;

__device__ __forceinline__ h4 ld4h(const float* __restrict__ p) {
  f4 v = *(const f4*)p;
  h4 r; r[0] = (_Float16)v[0]; r[1] = (_Float16)v[1];
  r[2] = (_Float16)v[2]; r[3] = (_Float16)v[3];
  return r;
}
__device__ __forceinline__ h4 ld4h(const _Float16* __restrict__ p) {
  return *(const h4*)p;
}

template<int ACT> __device__ __forceinline__ float activate(float x) {
  if constexpr (ACT == 1) return fmaxf(x, 0.0f);
  else if constexpr (ACT == 2) return tanhf(x);
  else return x;
}

// ---------------------------------------------------------------------------
// WMMA GEMM: C[M x N] = act(A[M x K] * B[N x K]^T + bias[N])
// f16 multiplicands (converted on LDS fill), f32 accumulate.
// Block tile 128x128x32, 256 threads = 8 waves in a 2(m) x 4(n) grid,
// each wave owns a 64x32 sub-tile = 4x2 fragments of 16x16.
// Double-buffered LDS; OOB rows/cols are index-CLAMPED (not zeroed): the
// resulting garbage only feeds D elements the epilogue never stores.
// K must be a multiple of 32 (true for all call sites: 512 / 4096).
// ---------------------------------------------------------------------------
#define BM 128
#define BN 128
#define BK 32

template<int ACT, typename TA, typename TB, typename TOUT>
__global__ __launch_bounds__(256) void gemm_bt(
    const TA* __restrict__ A, const TB* __restrict__ B,
    const float* __restrict__ bias, TOUT* __restrict__ C,
    int M, int N, int K)
{
  __shared__ __align__(16) _Float16 As[2][BM * BK];
  __shared__ __align__(16) _Float16 Bs[2][BN * BK];

  const int bm0  = blockIdx.y * BM;
  const int bn0  = blockIdx.x * BN;
  const int tid  = threadIdx.x;
  const int lane = tid & 31;
  const int wave = tid >> 5;
  const int wm   = wave >> 2;   // 0..1 : 64-row half
  const int wn   = wave & 3;    // 0..3 : 32-col quarter
  const int l16  = lane & 15;
  const int hi   = lane >> 4;   // lane half selects K-chunk / M-offset

  // Fill assignment: 4 h4-chunks per thread for each of A and B.
  int    lofs[4];              // LDS offset (halves)
  size_t aofs[4], bofs[4];     // global element offsets, rows clamped in-range
  #pragma unroll
  for (int p = 0; p < 4; ++p) {
    int e = tid + p * 256;
    int r = e >> 3, kc = (e & 7) << 2;   // BK/4 = 8 chunks per row
    lofs[p] = r * BK + kc;
    int ar = bm0 + r; if (ar > M - 1) ar = M - 1;
    int br = bn0 + r; if (br > N - 1) br = N - 1;
    aofs[p] = (size_t)ar * K + kc;
    bofs[p] = (size_t)br * K + kc;
  }

  v8f acc[4][2];
  #pragma unroll
  for (int mi = 0; mi < 4; ++mi)
    #pragma unroll
    for (int ni = 0; ni < 2; ++ni)
      #pragma unroll
      for (int v = 0; v < 8; ++v) acc[mi][ni][v] = 0.0f;

  // prologue: fill buffer 0
  h4 areg[4], breg[4];
  #pragma unroll
  for (int p = 0; p < 4; ++p) areg[p] = ld4h(&A[aofs[p]]);
  #pragma unroll
  for (int p = 0; p < 4; ++p) breg[p] = ld4h(&B[bofs[p]]);
  #pragma unroll
  for (int p = 0; p < 4; ++p) {
    *(h4*)&As[0][lofs[p]] = areg[p];
    *(h4*)&Bs[0][lofs[p]] = breg[p];
  }

  const int nk = K / BK;
  for (int t = 0; t < nk; ++t) {
    const int cur = t & 1;
    __syncthreads();           // buf[cur] ready; buf[cur^1] free to overwrite

    const bool more = (t + 1) < nk;
    if (more) {                // stage next tile while WMMAs run
      const int kt = (t + 1) * BK;
      #pragma unroll
      for (int p = 0; p < 4; ++p) areg[p] = ld4h(&A[aofs[p] + kt]);
      #pragma unroll
      for (int p = 0; p < 4; ++p) breg[p] = ld4h(&B[bofs[p] + kt]);
      if (t + 2 < nk) {        // warm GL2 two tiles ahead
        __builtin_prefetch(&A[aofs[0] + (size_t)(t + 2) * BK], 0, 0);
        __builtin_prefetch(&B[bofs[0] + (size_t)(t + 2) * BK], 0, 0);
      }
    }

    // --- fragments per ISA 7.12.2 layouts ----------------------------------
    // A 16x32 f16: lanes 0-15 row=l16 K[0..7]+K[16..23]; lanes 16-31 K[8..15]+K[24..31]
    v16h af[4];
    #pragma unroll
    for (int mi = 0; mi < 4; ++mi) {
      const _Float16* pa = &As[cur][(wm * 64 + mi * 16 + l16) * BK + hi * 8];
      ((h8*)&af[mi])[0] = *(const h8*)(pa);
      ((h8*)&af[mi])[1] = *(const h8*)(pa + 16);
    }
    // B 32x16 f16: lane = col; lanes 0-15 K[0..7]+K[16..23]; lanes 16-31 K[8..15]+K[24..31]
    v16h bf[2];
    #pragma unroll
    for (int ni = 0; ni < 2; ++ni) {
      const _Float16* pb = &Bs[cur][(wn * 32 + ni * 16 + l16) * BK + hi * 8];
      ((h8*)&bf[ni])[0] = *(const h8*)(pb);
      ((h8*)&bf[ni])[1] = *(const h8*)(pb + 16);
    }
    #pragma unroll
    for (int mi = 0; mi < 4; ++mi)
      #pragma unroll
      for (int ni = 0; ni < 2; ++ni)
        acc[mi][ni] = __builtin_amdgcn_wmma_f32_16x16x32_f16(
            false, af[mi], false, bf[ni], (short)0, acc[mi][ni], false, false);

    if (more) {                // commit staged tile to the alternate buffer
      #pragma unroll
      for (int p = 0; p < 4; ++p) {
        *(h4*)&As[cur ^ 1][lofs[p]] = areg[p];
        *(h4*)&Bs[cur ^ 1][lofs[p]] = breg[p];
      }
    }
  }

  // --- epilogue: D layout: lanes 0-15 M=v, lanes 16-31 M=8+v; N=l16 --------
  #pragma unroll
  for (int ni = 0; ni < 2; ++ni) {
    int col = bn0 + wn * 32 + ni * 16 + l16;
    if (col >= N) continue;
    float bv = bias ? bias[col] : 0.0f;
    #pragma unroll
    for (int mi = 0; mi < 4; ++mi) {
      #pragma unroll
      for (int v = 0; v < 8; ++v) {
        int row = bm0 + wm * 64 + mi * 16 + hi * 8 + v;
        if (row < M)
          C[(size_t)row * N + col] = (TOUT)activate<ACT>(acc[mi][ni][v] + bv);
      }
    }
  }
}

// ---------------------------------------------------------------------------
// Per-class masked-softmax attention. One block per class.
// attn[c] = (1/n) * sum_j w[j] * label0[idx_j],  w[j] = sum_i A[i,j]
// where A[i,:] = softmax over class members of gram[idx_i, idx_j].
// Deterministic: index list built by a single thread in class order.
// ---------------------------------------------------------------------------
__global__ __launch_bounds__(256) void attn_kernel(
    const float* __restrict__ gram, const _Float16* __restrict__ l0,
    const int* __restrict__ blab, float* __restrict__ attn,
    int* __restrict__ counts)
{
  __shared__ int   slab[Bsz];
  __shared__ int   sidx[Bsz];
  __shared__ float w[Bsz];
  __shared__ float red[256];
  __shared__ int   sn;

  const int c = blockIdx.x, tid = threadIdx.x;
  for (int b = tid; b < Bsz; b += 256) slab[b] = blab[b];
  __syncthreads();
  if (tid == 0) {
    int n = 0;
    for (int b = 0; b < Bsz; ++b) if (slab[b] == c) sidx[n++] = b;
    sn = n; counts[c] = n;
  }
  __syncthreads();
  const int n = sn;
  for (int j = tid; j < Bsz; j += 256) w[j] = 0.0f;
  __syncthreads();

  if (n == 0) {   // empty class: result masked out downstream anyway
    for (int f = tid; f < MFdim; f += 256) attn[(size_t)c * MFdim + f] = 0.0f;
    return;
  }

  for (int i = 0; i < n; ++i) {
    const float* g = gram + (size_t)sidx[i] * Bsz;
    // row max over class members
    float m = -3.4e38f;
    for (int j = tid; j < n; j += 256) m = fmaxf(m, g[sidx[j]]);
    red[tid] = m; __syncthreads();
    for (int s = 128; s > 0; s >>= 1) {
      if (tid < s) red[tid] = fmaxf(red[tid], red[tid + s]);
      __syncthreads();
    }
    m = red[0]; __syncthreads();
    // sum of exps
    float ssum = 0.0f;
    for (int j = tid; j < n; j += 256) ssum += __expf(g[sidx[j]] - m);
    red[tid] = ssum; __syncthreads();
    for (int s = 128; s > 0; s >>= 1) {
      if (tid < s) red[tid] += red[tid + s];
      __syncthreads();
    }
    float inv = 1.0f / red[0]; __syncthreads();
    for (int j = tid; j < n; j += 256) w[j] += __expf(g[sidx[j]] - m) * inv;
  }
  __syncthreads();

  const float invn = 1.0f / (float)n;
  for (int f = tid; f < MFdim; f += 256) {
    float acc = 0.0f;
    for (int j = 0; j < n; ++j)
      acc += w[j] * (float)l0[(size_t)sidx[j] * MFdim + f];
    attn[(size_t)c * MFdim + f] = acc * invn;
  }
}

// hs = nonempty ? sigmoid(t1 + t2) : hs_init   (t1/t2 already carry b7/b8)
__global__ __launch_bounds__(256) void hs_combine_kernel(
    const float* __restrict__ t1, const float* __restrict__ t2,
    const float* __restrict__ hsinit, const int* __restrict__ counts,
    float* __restrict__ hs)
{
  int i = blockIdx.x * 256 + threadIdx.x;
  if (i >= Cn * MFdim) return;
  int c = i >> 12;  // / MFdim
  hs[i] = (counts[c] > 0) ? 1.0f / (1.0f + __expf(-(t1[i] + t2[i]))) : hsinit[i];
}

// labels = nonempty ? tanh(t3) : labels_init
__global__ __launch_bounds__(256) void labels_combine_kernel(
    const float* __restrict__ t3, const float* __restrict__ linit,
    const int* __restrict__ counts, float* __restrict__ labels)
{
  int i = blockIdx.x * 256 + threadIdx.x;
  if (i >= Cn * MFdim) return;
  int c = i >> 12;
  labels[i] = (counts[c] > 0) ? tanhf(t3[i]) : linit[i];
}

__global__ __launch_bounds__(256) void lnorm_kernel(
    const float* __restrict__ labels, float* __restrict__ lnorm)
{
  __shared__ float red[256];
  const int c = blockIdx.x, tid = threadIdx.x;
  float s = 0.0f;
  for (int f = tid; f < MFdim; f += 256) {
    float v = labels[(size_t)c * MFdim + f];
    s += v * v;
  }
  red[tid] = s; __syncthreads();
  for (int sft = 128; sft > 0; sft >>= 1) {
    if (tid < sft) red[tid] += red[tid + sft];
    __syncthreads();
  }
  if (tid == 0) lnorm[c] = sqrtf(red[0]);
}

// outs[b,c] = (out_b . labels_c) / max(|out_b| * |labels_c|, 1e-8)
// 8 waves per block, one batch row per wave; labels staged in 40KB LDS chunks.
#define COSCH 512
__global__ __launch_bounds__(256) void cosine_kernel(
    const float* __restrict__ outm, const float* __restrict__ labels,
    const float* __restrict__ lnorm, float* __restrict__ dout)
{
  __shared__ float sl[Cn * COSCH];
  const int wave = threadIdx.x >> 5, lane = threadIdx.x & 31;
  const int b = blockIdx.x * 8 + wave;

  float dot[Cn];
  #pragma unroll
  for (int c = 0; c < Cn; ++c) dot[c] = 0.0f;
  float self = 0.0f;

  for (int f0 = 0; f0 < MFdim; f0 += COSCH) {
    __syncthreads();
    for (int i = threadIdx.x; i < Cn * COSCH; i += 256) {
      int c = i / COSCH, f = i % COSCH;
      sl[i] = labels[(size_t)c * MFdim + f0 + f];
    }
    __syncthreads();
    for (int f = lane; f < COSCH; f += 32) {
      float o = outm[(size_t)b * MFdim + f0 + f];
      self += o * o;
      #pragma unroll
      for (int c = 0; c < Cn; ++c) dot[c] += o * sl[c * COSCH + f];
    }
  }
  #pragma unroll
  for (int s = 16; s > 0; s >>= 1) {
    self += __shfl_down(self, s, 32);
    #pragma unroll
    for (int c = 0; c < Cn; ++c) dot[c] += __shfl_down(dot[c], s, 32);
  }
  if (lane == 0) {
    float on = sqrtf(self);
    #pragma unroll
    for (int c = 0; c < Cn; ++c)
      dout[(size_t)b * Cn + c] = dot[c] / fmaxf(on * lnorm[c], 1e-8f);
  }
}

// ---------------------------------------------------------------------------
extern "C" void kernel_launch(void* const* d_in, const int* in_sizes, int n_in,
                              void* d_out, int out_size, void* d_ws, size_t ws_size,
                              hipStream_t stream) {
  (void)in_sizes; (void)n_in; (void)out_size; (void)ws_size;
  const float* x      = (const float*)d_in[0];
  const int*   blab   = (const int*)  d_in[1];
  const float* W1 = (const float*)d_in[2];  const float* b1 = (const float*)d_in[3];
  const float* W2 = (const float*)d_in[4];  const float* b2 = (const float*)d_in[5];
  const float* W3 = (const float*)d_in[6];  const float* b3 = (const float*)d_in[7];
  const float* W4 = (const float*)d_in[8];  const float* b4 = (const float*)d_in[9];
  const float* W5 = (const float*)d_in[10]; const float* b5 = (const float*)d_in[11];
  const float* W7 = (const float*)d_in[12]; const float* b7 = (const float*)d_in[13];
  const float* W8 = (const float*)d_in[14]; const float* b8 = (const float*)d_in[15];
  const float* W9 = (const float*)d_in[16]; const float* b9 = (const float*)d_in[17];
  const float* hsinit = (const float*)d_in[18];
  const float* linit  = (const float*)d_in[19];
  float* out = (float*)d_out;

  // workspace bump allocator (total ~78 MB)
  char* wptr = (char*)d_ws;
  auto alloc = [&](size_t bytes) {
    char* p = wptr; wptr += (bytes + 255) & ~(size_t)255; return p;
  };
  float*    h1      = (float*)   alloc((size_t)Bsz * Fdim  * 4);
  float*    h2      = (float*)   alloc((size_t)Bsz * Fdim  * 4);
  _Float16* l0h     = (_Float16*)alloc((size_t)Bsz * MFdim * 2);
  float*    outm    = (float*)   alloc((size_t)Bsz * MFdim * 4);
  float*    gram    = (float*)   alloc((size_t)Bsz * Bsz   * 4);
  float*    attn    = (float*)   alloc((size_t)Cn  * MFdim * 4);
  float*    t1      = (float*)   alloc((size_t)Cn  * MFdim * 4);
  float*    t2      = (float*)   alloc((size_t)Cn  * MFdim * 4);
  float*    hs      = (float*)   alloc((size_t)Cn  * MFdim * 4);
  float*    t3      = (float*)   alloc((size_t)Cn  * MFdim * 4);
  float*    labelsM = (float*)   alloc((size_t)Cn  * MFdim * 4);
  int*      counts  = (int*)     alloc((size_t)Cn * 4);
  float*    lnorm   = (float*)   alloc((size_t)Cn * 4);

  const dim3 blk(256);
  auto grd = [](int M, int N) { return dim3((N + BN - 1) / BN, (M + BM - 1) / BM); };

  // MLP trunk
  gemm_bt<1, float, float, float><<<grd(Bsz, Fdim), blk, 0, stream>>>(
      x,  W1, b1, h1, Bsz, Fdim, Fdim);
  gemm_bt<1, float, float, float><<<grd(Bsz, Fdim), blk, 0, stream>>>(
      h1, W2, b2, h2, Bsz, Fdim, Fdim);
  gemm_bt<2, float, float, _Float16><<<grd(Bsz, MFdim), blk, 0, stream>>>(
      h2, W3, b3, l0h, Bsz, MFdim, Fdim);
  gemm_bt<2, float, float, float><<<grd(Bsz, MFdim), blk, 0, stream>>>(
      h2, W4, b4, outm, Bsz, MFdim, Fdim);

  // Gram matrix: label0 . label0^T (f16 in, f32 accumulate)
  gemm_bt<0, _Float16, _Float16, float><<<grd(Bsz, Bsz), blk, 0, stream>>>(
      l0h, l0h, (const float*)nullptr, gram, Bsz, Bsz, MFdim);

  // per-class masked softmax attention
  attn_kernel<<<dim3(Cn), blk, 0, stream>>>(gram, l0h, blab, attn, counts);

  // hs = sigmoid(attn W7^T + b7 + hs_init W8^T + b8), masked by nonempty
  gemm_bt<0, float, float, float><<<grd(Cn, MFdim), blk, 0, stream>>>(
      attn,   W7, b7, t1, Cn, MFdim, MFdim);
  gemm_bt<0, float, float, float><<<grd(Cn, MFdim), blk, 0, stream>>>(
      hsinit, W8, b8, t2, Cn, MFdim, MFdim);
  hs_combine_kernel<<<dim3((Cn * MFdim + 255) / 256), blk, 0, stream>>>(
      t1, t2, hsinit, counts, hs);

  // labels = tanh(hs W9^T + b9), masked by nonempty
  gemm_bt<0, float, float, float><<<grd(Cn, MFdim), blk, 0, stream>>>(
      hs, W9, b9, t3, Cn, MFdim, MFdim);
  labels_combine_kernel<<<dim3((Cn * MFdim + 255) / 256), blk, 0, stream>>>(
      t3, linit, counts, labelsM);

  // cosine similarity block -> d_out rows [0, 2048)
  lnorm_kernel<<<dim3(Cn), blk, 0, stream>>>(labelsM, lnorm);
  cosine_kernel<<<dim3(Bsz / 8), blk, 0, stream>>>(outm, labelsM, lnorm, out);

  // ls = tanh(labels W5^T + b5) -> d_out rows [2048, 2068)
  gemm_bt<2, float, float, float><<<grd(Cn, Cn), blk, 0, stream>>>(
      labelsM, W5, b5, out + (size_t)Bsz * Cn, Cn, Cn, MFdim);
}
